// GraphConvEncoder_30562987278567
// MI455X (gfx1250) — compile-verified, matrix operand order checked by reading, not run on hardware
//
#include <hip/hip_runtime.h>
#include <hip/hip_bf16.h>

typedef __attribute__((ext_vector_type(16))) __bf16 v16bf;
typedef __attribute__((ext_vector_type(8)))  __bf16 v8bf;
typedef __attribute__((ext_vector_type(8)))  float  v8f;

#define CAT16(lo, hi) __builtin_shufflevector(lo, hi, 0,1,2,3,4,5,6,7,8,9,10,11,12,13,14,15)

static __device__ __forceinline__ void atomAddF32(float* p, float v) {
    unsafeAtomicAdd(p, v);   // hardware global_atomic_add_f32 (order-insensitive sum)
}

// ---------------- zero scratch (float4 stores) ----------------
__global__ void k_zero(float4* __restrict__ p, long n4) {
    long i = (long)blockIdx.x * blockDim.x + threadIdx.x;
    if (i < n4) p[i] = make_float4(0.f, 0.f, 0.f, 0.f);
}

// ---------------- weights: f32 row-major [K][N] -> bf16 transposed [N][K] ----------------
__global__ void k_prep_weights(const float* __restrict__ w1_rel, const float* __restrict__ w1_root,
                               const float* __restrict__ w2_rel, const float* __restrict__ w2_root,
                               __bf16* __restrict__ w1relT, __bf16* __restrict__ w1rootT,
                               __bf16* __restrict__ w2relT, __bf16* __restrict__ w2rootT) {
    int tid = blockIdx.x * blockDim.x + threadIdx.x;     // 0 .. 128*128-1
    int k = tid >> 7, n = tid & 127;
    w1relT [n * 128 + k] = (__bf16)w1_rel [k * 128 + n];
    w1rootT[n * 128 + k] = (__bf16)w1_root[k * 128 + n];
    if (n < 64) {
        w2relT [n * 128 + k] = (__bf16)w2_rel [k * 64 + n];
        w2rootT[n * 128 + k] = (__bf16)w2_root[k * 64 + n];
    }
}

// ---------------- scatter layer 1: agg[dst] += x[src], 128 f32/edge, 1 wave per edge ----------------
__global__ __launch_bounds__(256) void k_scatter128(const float* __restrict__ x,
                                                    const int* __restrict__ eidx,
                                                    float* __restrict__ agg, int E) {
    int e = blockIdx.x * 8 + (threadIdx.x >> 5);
    if (e >= E) return;
    int lane = threadIdx.x & 31;
    int s = eidx[e], d = eidx[E + e];
    float4 v = ((const float4*)(x + (size_t)s * 128))[lane];
    float* dp = agg + (size_t)d * 128 + lane * 4;
    atomAddF32(dp + 0, v.x); atomAddF32(dp + 1, v.y);
    atomAddF32(dp + 2, v.z); atomAddF32(dp + 3, v.w);
}

// ---------------- A-fragment builder from padded LDS tile (16x32 bf16 per k-step) ----------------
static __device__ __forceinline__ v16bf frag_a(const __bf16* rowPtr, int k0, int half) {
    v8bf lo = *(const v8bf*)(rowPtr + k0 + half * 8);
    v8bf hi = *(const v8bf*)(rowPtr + k0 + 16 + half * 8);
    return CAT16(lo, hi);
}
static __device__ __forceinline__ v16bf frag_b(const __bf16* __restrict__ wT, int nc, int k0, int half) {
    v8bf lo = *(const v8bf*)(wT + (size_t)nc * 128 + k0 + half * 8);
    v8bf hi = *(const v8bf*)(wT + (size_t)nc * 128 + k0 + 16 + half * 8);
    return CAT16(lo, hi);
}

// ---------------- layer 1: h = relu(agg@W1_rel + b1 + x@W1_root), h stored bf16 ----------------
__global__ __launch_bounds__(256) void k_gemm1(const float* __restrict__ agg,
                                               const float* __restrict__ x,
                                               const __bf16* __restrict__ wrelT,
                                               const __bf16* __restrict__ wrootT,
                                               const float* __restrict__ b1,
                                               __bf16* __restrict__ h, int N) {
    __shared__ __bf16 Aagg[16][136];   // +8 pad: spread LDS banks
    __shared__ __bf16 Ax[16][136];
    const int rowBase = blockIdx.x * 16;
    const int tid = threadIdx.x;
    for (int idx = tid; idx < 16 * 128; idx += 256) {
        int r = idx >> 7, c = idx & 127;
        bool ok = (rowBase + r) < N;
        size_t g = (size_t)(rowBase + r) * 128 + c;
        Aagg[r][c] = ok ? (__bf16)agg[g] : (__bf16)0.f;
        Ax[r][c]   = ok ? (__bf16)x[g]   : (__bf16)0.f;
    }
    __syncthreads();

    const int wave = tid >> 5, lane = tid & 31;
    const int row = lane & 15, half = lane >> 4;
    const int nc = wave * 16 + row;          // output / B column
    v8f c = {};
#pragma unroll
    for (int kb = 0; kb < 4; ++kb) {
        const int k0 = kb * 32;
        v16bf aA = frag_a(&Aagg[row][0], k0, half);
        v16bf bR = frag_b(wrelT, nc, k0, half);
        c = __builtin_amdgcn_wmma_f32_16x16x32_bf16(false, aA, false, bR, (short)0, c, false, false);
        v16bf aX = frag_a(&Ax[row][0], k0, half);
        v16bf bT = frag_b(wrootT, nc, k0, half);
        c = __builtin_amdgcn_wmma_f32_16x16x32_bf16(false, aX, false, bT, (short)0, c, false, false);
    }
    const float bias = b1[nc];
#pragma unroll
    for (int i = 0; i < 8; ++i) {
        int rr = rowBase + half * 8 + i;     // D layout: M = half*8 + vgpr
        if (rr < N) {
            float v = c[i] + bias;
            h[(size_t)rr * 128 + nc] = (__bf16)(v > 0.f ? v : 0.f);
        }
    }
}

// ---------------- layer 2: waves 0-3 -> y2 = h@W2_rel ; waves 4-7 -> out = h@W2_root + b2 ----------------
__global__ __launch_bounds__(256) void k_gemm2(const __bf16* __restrict__ h,
                                               const __bf16* __restrict__ w2relT,
                                               const __bf16* __restrict__ w2rootT,
                                               const float* __restrict__ b2,
                                               float* __restrict__ y2,
                                               float* __restrict__ out, int N) {
    __shared__ __bf16 Ah[16][136];
    const int rowBase = blockIdx.x * 16;
    const int tid = threadIdx.x;
    for (int idx = tid; idx < 16 * 128; idx += 256) {
        int r = idx >> 7, c = idx & 127;
        Ah[r][c] = (rowBase + r) < N ? h[(size_t)(rowBase + r) * 128 + c] : (__bf16)0.f;
    }
    __syncthreads();

    const int wave = tid >> 5, lane = tid & 31;
    const int row = lane & 15, half = lane >> 4;
    const bool isRoot = wave >= 4;
    const int nc = (wave & 3) * 16 + row;            // 0..63
    const __bf16* wT = isRoot ? w2rootT : w2relT;
    v8f c = {};
#pragma unroll
    for (int kb = 0; kb < 4; ++kb) {
        const int k0 = kb * 32;
        v16bf a = frag_a(&Ah[row][0], k0, half);
        v16bf b = frag_b(wT, nc, k0, half);
        c = __builtin_amdgcn_wmma_f32_16x16x32_bf16(false, a, false, b, (short)0, c, false, false);
    }
    if (isRoot) {
        const float bias = b2[nc];
#pragma unroll
        for (int i = 0; i < 8; ++i) {
            int rr = rowBase + half * 8 + i;
            if (rr < N) out[(size_t)rr * 64 + nc] = c[i] + bias;
        }
    } else {
#pragma unroll
        for (int i = 0; i < 8; ++i) {
            int rr = rowBase + half * 8 + i;
            if (rr < N) y2[(size_t)rr * 64 + nc] = c[i];
        }
    }
}

// ---------------- scatter layer 2: out[dst] += y2[src], 64 f32/edge, 1 wave per edge ----------------
__global__ __launch_bounds__(256) void k_scatter64(const float* __restrict__ y2,
                                                   const int* __restrict__ eidx,
                                                   float* __restrict__ out, int E) {
    int e = blockIdx.x * 8 + (threadIdx.x >> 5);
    if (e >= E) return;
    int lane = threadIdx.x & 31;
    int s = eidx[e], d = eidx[E + e];
    float2 v = ((const float2*)(y2 + (size_t)s * 64))[lane];
    float* dp = out + (size_t)d * 64 + lane * 2;
    atomAddF32(dp + 0, v.x);
    atomAddF32(dp + 1, v.y);
}

extern "C" void kernel_launch(void* const* d_in, const int* in_sizes, int n_in,
                              void* d_out, int out_size, void* d_ws, size_t ws_size,
                              hipStream_t stream) {
    const float* x       = (const float*)d_in[0];
    const int*   eidx    = (const int*)d_in[1];
    const float* w1_rel  = (const float*)d_in[2];
    const float* b1      = (const float*)d_in[3];
    const float* w1_root = (const float*)d_in[4];
    const float* w2_rel  = (const float*)d_in[5];
    const float* b2      = (const float*)d_in[6];
    const float* w2_root = (const float*)d_in[7];
    float* out = (float*)d_out;

    const int N = in_sizes[0] / 128;
    const int E = in_sizes[1] / 2;

    char* ws = (char*)d_ws;
    size_t off = 0;
    auto take = [&](size_t bytes) -> void* {
        void* p = ws + off;
        off += (bytes + 255) & ~(size_t)255;
        return p;
    };
    float*  agg1    = (float*)take((size_t)N * 128 * sizeof(float));
    __bf16* h       = (__bf16*)take((size_t)N * 128 * sizeof(__bf16));
    float*  y2      = (float*)take((size_t)N * 64 * sizeof(float));
    __bf16* w1relT  = (__bf16*)take(128 * 128 * sizeof(__bf16));
    __bf16* w1rootT = (__bf16*)take(128 * 128 * sizeof(__bf16));
    __bf16* w2relT  = (__bf16*)take(64 * 128 * sizeof(__bf16));
    __bf16* w2rootT = (__bf16*)take(64 * 128 * sizeof(__bf16));

    // 1) weight transpose + bf16 convert
    k_prep_weights<<<64, 256, 0, stream>>>(w1_rel, w1_root, w2_rel, w2_root,
                                           w1relT, w1rootT, w2relT, w2rootT);
    // 2) zero agg1
    long n4 = (long)N * 128 / 4;
    k_zero<<<(int)((n4 + 255) / 256), 256, 0, stream>>>((float4*)agg1, n4);
    // 3) edge scatter of raw features
    int eBlocks = (E + 7) / 8;
    k_scatter128<<<eBlocks, 256, 0, stream>>>(x, eidx, agg1, E);
    // 4) layer-1 fused GEMM (+bias+ReLU), h in bf16
    int rBlocks = (N + 15) / 16;
    k_gemm1<<<rBlocks, 256, 0, stream>>>(agg1, x, w1relT, w1rootT, b1, h, N);
    // 5) layer-2 GEMMs: y2 = h@W2_rel (scratch), out = h@W2_root + b2 (direct)
    k_gemm2<<<rBlocks, 256, 0, stream>>>(h, w2relT, w2rootT, b2, y2, out, N);
    // 6) edge scatter of transformed features (64-wide, half the traffic of pre-transform)
    k_scatter64<<<eBlocks, 256, 0, stream>>>(y2, eidx, out, E);
}
